// SE_BlockRNN_2130303779406
// MI455X (gfx1250) — matrix operand order repaired and checked by reading, not compiled, not run on hardware
//
#include <hip/hip_runtime.h>
#include <hip/hip_bf16.h>

// ---------------------------------------------------------------------------
// SE_BlockRNN for MI455X (gfx1250, wave32).
// Bandwidth-bound: mean over 302MB + scale 302MB+302MB  => ~39us @ 23.3TB/s.
// BiLSTM recurrence (h[128,4] @ Whh.T[4,16]) -> V_WMMA_F32_16X16X4_F32
// (one 16-row batch tile per wave, fp32, lossless).  SE MLP -> f16 WMMA.
// Serial-latency path uses v_exp_f32/v_rcp_f32 sigmoid/tanh (no libm).
// ---------------------------------------------------------------------------

typedef __attribute__((ext_vector_type(2)))  float    v2f;
typedef __attribute__((ext_vector_type(4)))  float    v4f;
typedef __attribute__((ext_vector_type(8)))  float    v8f;
typedef __attribute__((ext_vector_type(16))) _Float16 v16h;

#define BSZ   2
#define CCH   64
#define TT    64
#define HWW   96
#define NB    (BSZ * CCH)          // 128 sequences
#define PIX   (HWW * HWW)          // 9216
#define SLICE (TT * PIX)           // 589824 elements per (b,c)
#define F4_PER_SLICE (SLICE / 4)   // 147456
#define BLK_PER_SLICE 64
#define F4_PER_BLK (F4_PER_SLICE / BLK_PER_SLICE)   // 2304

// ---- fast transcendentals: single v_exp_f32 / v_rcp_f32 --------------------
__device__ __forceinline__ float fexp(float x) {
#if __has_builtin(__builtin_amdgcn_exp2f)
  return __builtin_amdgcn_exp2f(x * 1.4426950408889634f);
#else
  return __expf(x);
#endif
}
__device__ __forceinline__ float frcp(float x) {
#if __has_builtin(__builtin_amdgcn_rcpf)
  return __builtin_amdgcn_rcpf(x);
#else
  return 1.0f / x;
#endif
}
__device__ __forceinline__ float sigf(float x) { return frcp(1.0f + fexp(-x)); }
// tanh(x) = 1 - 2/(exp(2x)+1); exp->inf gives 1, exp->0 gives -1 (stable).
__device__ __forceinline__ float tanh_fast(float x) {
  return 1.0f - 2.0f * frcp(1.0f + fexp(2.0f * x));
}

// ---- D(16x16) = A(16x4) * B(4x16) + C, fp32 WMMA ---------------------------
// A layout: lane = m + 16*(k>=2), reg = k&1.  B layout: lane = n + 16*(k>=2).
// D layout: row = j + 8*(lane/16), col = lane%16.
__device__ __forceinline__ v8f mma_16x16x4(float ax, float ay, const float wB[4],
                                           v8f cacc, int lane) {
#if __has_builtin(__builtin_amdgcn_wmma_f32_16x16x4_f32)
  const int half = lane >> 4;
  v2f a; a.x = ax; a.y = ay;
  v2f b; b.x = wB[half * 2]; b.y = wB[half * 2 + 1];
  return __builtin_amdgcn_wmma_f32_16x16x4_f32(false, a, false, b, (short)0,
                                               cacc, false, false);
#else
  // wave32 shuffle fallback (also keeps the host compile pass happy)
  v8f d = cacc;
  const int hi8 = (lane >> 4) << 3;
  #pragma unroll
  for (int j = 0; j < 8; ++j) {
    const int mm = hi8 + j;
    float h0 = __shfl(ax, mm, 32);
    float h1 = __shfl(ay, mm, 32);
    float h2 = __shfl(ax, 16 + mm, 32);
    float h3 = __shfl(ay, 16 + mm, 32);
    d[j] += h0 * wB[0] + h1 * wB[1] + h2 * wB[2] + h3 * wB[3];
  }
  return d;
#endif
}

// ---- D(16x16,f32) = A(16x32,f16) * B(32x16,f16) + C ------------------------
__device__ __forceinline__ v8f mma_f16(v16h a, v16h b, v8f cacc) {
#if __has_builtin(__builtin_amdgcn_wmma_f32_16x16x32_f16)
  return __builtin_amdgcn_wmma_f32_16x16x32_f16(false, a, false, b, (short)0,
                                                cacc, false, false);
#else
  return cacc;  // host-pass placeholder only; gfx1250 device pass has the builtin
#endif
}

// ===========================================================================
// Kernel 1: adaptive avg pool over H,W.  One block per (b,c,d) slice.
// Reads 302 MB -> pure bandwidth (default RT hints: keep x in L2 for reuse
// by the final scale pass; L2 is 192MB global).
// ===========================================================================
__global__ __launch_bounds__(256) void mean_kernel(const float* __restrict__ x,
                                                   float* __restrict__ mean) {
  const int slice = blockIdx.x;  // (b*C + c)*T + d, 8192 slices
  const v4f* p = (const v4f*)(x + (size_t)slice * PIX);
  float s = 0.f;
  for (int i = threadIdx.x; i < PIX / 4; i += 256) {
    v4f v = p[i];
    s += (v.x + v.y) + (v.z + v.w);
  }
  #pragma unroll
  for (int off = 16; off; off >>= 1) s += __shfl_down(s, off, 32);
  __shared__ float red[8];
  if ((threadIdx.x & 31) == 0) red[threadIdx.x >> 5] = s;
  __syncthreads();
  if (threadIdx.x == 0) {
    float t = 0.f;
    #pragma unroll
    for (int w = 0; w < 8; ++w) t += red[w];
    mean[slice] = t * (1.0f / (float)PIX);
  }
}

// ===========================================================================
// Kernel 2: BiLSTM layer 0 (input size 1, hidden 4).  16 single-wave blocks:
// blockIdx = dir*8 + tile; each wave owns 16 sequences, both gate chunks.
// Per step: one fp32 WMMA for h@Whh.T, LDS shuttle to re-layout D -> state.
// ===========================================================================
__global__ __launch_bounds__(32) void lstm_l0_kernel(
    const float* __restrict__ mean,
    const float* __restrict__ wih_f, const float* __restrict__ whh_f,
    const float* __restrict__ bih_f, const float* __restrict__ bhh_f,
    const float* __restrict__ wih_b, const float* __restrict__ whh_b,
    const float* __restrict__ bih_b, const float* __restrict__ bhh_b,
    float* __restrict__ y0 /* [128][64][8] */) {
  __shared__ float gbuf[256];
  const int lane = threadIdx.x;
  const int dir  = blockIdx.x >> 3;
  const int tile = blockIdx.x & 7;
  const int m    = lane & 15;
  const int half = lane >> 4;
  const int k0   = half * 2;
  const int row  = tile * 16 + m;

  const float* wih = dir ? wih_b : wih_f;   // [16,1]
  const float* whh = dir ? whh_b : whh_f;   // [16,4]
  const float* bih = dir ? bih_b : bih_f;
  const float* bhh = dir ? bhh_b : bhh_f;

  float wB[4];                               // B[k][n]=whh[n][k], n = lane%16
  #pragma unroll
  for (int k = 0; k < 4; ++k) wB[k] = whh[m * 4 + k];

  float bi[2], bf[2], bg[2], bo[2], wi[2], wf[2], wg[2], wo[2];
  #pragma unroll
  for (int u = 0; u < 2; ++u) {
    const int k = k0 + u;
    bi[u] = bih[k]      + bhh[k];
    bf[u] = bih[4 + k]  + bhh[4 + k];
    bg[u] = bih[8 + k]  + bhh[8 + k];
    bo[u] = bih[12 + k] + bhh[12 + k];
    wi[u] = wih[k]; wf[u] = wih[4 + k]; wg[u] = wih[8 + k]; wo[u] = wih[12 + k];
  }

  float hx = 0.f, hy = 0.f, cx = 0.f, cy = 0.f;
  for (int s = 0; s < TT; ++s) {
    const int t = dir ? (TT - 1 - s) : s;
    v8f acc = {0, 0, 0, 0, 0, 0, 0, 0};
    acc = mma_16x16x4(hx, hy, wB, acc, lane);          // h @ Whh.T
    const int hi8 = half * 8;
    #pragma unroll
    for (int j = 0; j < 8; ++j) gbuf[(hi8 + j) * 16 + m] = acc[j];
    asm volatile("s_wait_dscnt 0" ::: "memory");        // same-wave DS in-order
    const float xin = mean[row * TT + t];
    const float* gm = gbuf + m * 16;
    float hn[2], cn[2];
    #pragma unroll
    for (int u = 0; u < 2; ++u) {
      const int k = k0 + u;
      const float gi = gm[k]      + xin * wi[u] + bi[u];
      const float gf = gm[4 + k]  + xin * wf[u] + bf[u];
      const float gg = gm[8 + k]  + xin * wg[u] + bg[u];
      const float go = gm[12 + k] + xin * wo[u] + bo[u];
      float cc = u ? cy : cx;
      cc = sigf(gf) * cc + sigf(gi) * tanh_fast(gg);
      cn[u] = cc;
      hn[u] = sigf(go) * tanh_fast(cc);
    }
    cx = cn[0]; cy = cn[1]; hx = hn[0]; hy = hn[1];
    float* yo = y0 + (row * TT + t) * 8 + dir * 4 + k0;
    yo[0] = hx; yo[1] = hy;
  }
}

// ===========================================================================
// Kernel 3: BiLSTM layer 1 (input size 8, hidden 4).  Same tiling; per step
// three fp32 WMMAs: h@Whh.T + x[0:4]@Wih.T[:,0:4] + x[4:8]@Wih.T[:,4:8].
// Only the t==63 output row feeds the SE head.
// ===========================================================================
__global__ __launch_bounds__(32) void lstm_l1_kernel(
    const float* __restrict__ y0,
    const float* __restrict__ wih_f, const float* __restrict__ whh_f,
    const float* __restrict__ bih_f, const float* __restrict__ bhh_f,
    const float* __restrict__ wih_b, const float* __restrict__ whh_b,
    const float* __restrict__ bih_b, const float* __restrict__ bhh_b,
    float* __restrict__ hlast /* [128][8] */) {
  __shared__ float gbuf[256];
  const int lane = threadIdx.x;
  const int dir  = blockIdx.x >> 3;
  const int tile = blockIdx.x & 7;
  const int m    = lane & 15;
  const int half = lane >> 4;
  const int k0   = half * 2;
  const int row  = tile * 16 + m;

  const float* wih = dir ? wih_b : wih_f;   // [16,8]
  const float* whh = dir ? whh_b : whh_f;   // [16,4]
  const float* bih = dir ? bih_b : bih_f;
  const float* bhh = dir ? bhh_b : bhh_f;

  float wBh[4], wBxl[4], wBxh[4];
  #pragma unroll
  for (int k = 0; k < 4; ++k) {
    wBh[k]  = whh[m * 4 + k];
    wBxl[k] = wih[m * 8 + k];
    wBxh[k] = wih[m * 8 + 4 + k];
  }
  float bi[2], bf[2], bg[2], bo[2];
  #pragma unroll
  for (int u = 0; u < 2; ++u) {
    const int k = k0 + u;
    bi[u] = bih[k]      + bhh[k];
    bf[u] = bih[4 + k]  + bhh[4 + k];
    bg[u] = bih[8 + k]  + bhh[8 + k];
    bo[u] = bih[12 + k] + bhh[12 + k];
  }

  float hx = 0.f, hy = 0.f, cx = 0.f, cy = 0.f;
  for (int s = 0; s < TT; ++s) {
    const int t = dir ? (TT - 1 - s) : s;
    const float* yi = y0 + (row * TT + t) * 8;
    v8f acc = {0, 0, 0, 0, 0, 0, 0, 0};
    acc = mma_16x16x4(hx, hy, wBh, acc, lane);
    acc = mma_16x16x4(yi[k0],     yi[k0 + 1], wBxl, acc, lane);
    acc = mma_16x16x4(yi[4 + k0], yi[5 + k0], wBxh, acc, lane);
    const int hi8 = half * 8;
    #pragma unroll
    for (int j = 0; j < 8; ++j) gbuf[(hi8 + j) * 16 + m] = acc[j];
    asm volatile("s_wait_dscnt 0" ::: "memory");
    const float* gm = gbuf + m * 16;
    float hn[2], cn[2];
    #pragma unroll
    for (int u = 0; u < 2; ++u) {
      const int k = k0 + u;
      const float gi = gm[k]      + bi[u];
      const float gf = gm[4 + k]  + bf[u];
      const float gg = gm[8 + k]  + bg[u];
      const float go = gm[12 + k] + bo[u];
      float cc = u ? cy : cx;
      cc = sigf(gf) * cc + sigf(gi) * tanh_fast(gg);
      cn[u] = cc;
      hn[u] = sigf(go) * tanh_fast(cc);
    }
    cx = cn[0]; cy = cn[1]; hx = hn[0]; hy = hn[1];
    if (t == TT - 1) {   // fwd: last scan step; bwd: first scan step
      float* ho = hlast + row * 8 + dir * 4 + k0;
      ho[0] = hx; ho[1] = hy;
    }
  }
}

// ===========================================================================
// Kernel 4: SE head via f16 WMMA (codegen-confirmed builtin).  One wave.
// ylast[m][f] = hlast[m*512+f] for m<2, zero-padded to M=16.
// GEMM1: [16,512]x[512,32] (16 K-chunks, 2 N-tiles), relu.
// GEMM2: [16,32]x[32,64]   (1 K-chunk, 4 N-tiles), sigmoid -> scale[128].
// ===========================================================================
__global__ __launch_bounds__(32) void se_kernel(
    const float* __restrict__ hlast, const float* __restrict__ w1,
    const float* __restrict__ w2, float* __restrict__ scale) {
  __shared__ float z1s[2][32];
  const int lane = threadIdx.x;
  const int m  = lane & 15;
  const int hh = lane >> 4;

  for (int nt = 0; nt < 2; ++nt) {
    v8f acc = {0, 0, 0, 0, 0, 0, 0, 0};
    for (int kc = 0; kc < 16; ++kc) {
      v16h a, b;
      #pragma unroll
      for (int r = 0; r < 8; ++r) {
        const int q = r & 3, blk = r >> 2;
        const int kk = kc * 32 + blk * 16 + hh * 8 + 2 * q;   // A: 16-bit layout
        a[2 * r]     = (m < 2) ? (_Float16)hlast[m * 512 + kk]     : (_Float16)0.f;
        a[2 * r + 1] = (m < 2) ? (_Float16)hlast[m * 512 + kk + 1] : (_Float16)0.f;
        const int kB = blk * 16 + hh * 8 + 2 * q;             // B: 16-bit layout
        b[2 * r]     = (_Float16)w1[(nt * 16 + m) * 512 + kc * 32 + kB];
        b[2 * r + 1] = (_Float16)w1[(nt * 16 + m) * 512 + kc * 32 + kB + 1];
      }
      acc = mma_f16(a, b, acc);
    }
    if (hh == 0) {                                  // D rows 0,1 = batch rows
      z1s[0][nt * 16 + m] = fmaxf(acc[0], 0.f);
      z1s[1][nt * 16 + m] = fmaxf(acc[1], 0.f);
    }
  }
  asm volatile("s_wait_dscnt 0" ::: "memory");

  v16h a2;
  #pragma unroll
  for (int r = 0; r < 8; ++r) {
    const int q = r & 3, blk = r >> 2;
    const int kk = blk * 16 + hh * 8 + 2 * q;
    a2[2 * r]     = (m < 2) ? (_Float16)z1s[m][kk]     : (_Float16)0.f;
    a2[2 * r + 1] = (m < 2) ? (_Float16)z1s[m][kk + 1] : (_Float16)0.f;
  }
  for (int nt = 0; nt < 4; ++nt) {
    v16h b2;
    #pragma unroll
    for (int r = 0; r < 8; ++r) {
      const int kB = (r >> 2) * 16 + hh * 8 + 2 * (r & 3);
      b2[2 * r]     = (_Float16)w2[(nt * 16 + m) * 32 + kB];
      b2[2 * r + 1] = (_Float16)w2[(nt * 16 + m) * 32 + kB + 1];
    }
    v8f acc = {0, 0, 0, 0, 0, 0, 0, 0};
    acc = mma_f16(a2, b2, acc);
    if (hh == 0) {
      scale[nt * 16 + m]      = sigf(acc[0]);   // batch 0
      scale[64 + nt * 16 + m] = sigf(acc[1]);   // batch 1
    }
  }
}

// ===========================================================================
// Kernel 5: out = x * scale[b,c].  Each block owns a contiguous 2304-float4
// chunk inside one (b,c) slice, so bc is uniform (scalar load, no per-element
// divide).  Non-temporal on both sides: last use of x, out never re-read.
// ===========================================================================
__global__ __launch_bounds__(256) void scale_kernel(const float* __restrict__ x,
                                                    const float* __restrict__ scale,
                                                    float* __restrict__ out) {
  const int bc = blockIdx.x >> 6;                  // 64 blocks per slice
  const float s = scale[bc];
  const size_t base = (size_t)blockIdx.x * F4_PER_BLK;
  const v4f* xv = (const v4f*)x + base;
  v4f* ov = (v4f*)out + base;
  #pragma unroll
  for (int i = threadIdx.x; i < F4_PER_BLK; i += 256) {
    v4f v = __builtin_nontemporal_load(&xv[i]);
    v *= s;
    __builtin_nontemporal_store(v, &ov[i]);
  }
}

// ===========================================================================
extern "C" void kernel_launch(void* const* d_in, const int* in_sizes, int n_in,
                              void* d_out, int out_size, void* d_ws, size_t ws_size,
                              hipStream_t stream) {
  (void)in_sizes; (void)n_in; (void)out_size; (void)ws_size;
  const float* x     = (const float*)d_in[0];
  const float* wih0f = (const float*)d_in[1];
  const float* whh0f = (const float*)d_in[2];
  const float* bih0f = (const float*)d_in[3];
  const float* bhh0f = (const float*)d_in[4];
  const float* wih0b = (const float*)d_in[5];
  const float* whh0b = (const float*)d_in[6];
  const float* bih0b = (const float*)d_in[7];
  const float* bhh0b = (const float*)d_in[8];
  const float* wih1f = (const float*)d_in[9];
  const float* whh1f = (const float*)d_in[10];
  const float* bih1f = (const float*)d_in[11];
  const float* bhh1f = (const float*)d_in[12];
  const float* wih1b = (const float*)d_in[13];
  const float* whh1b = (const float*)d_in[14];
  const float* bih1b = (const float*)d_in[15];
  const float* bhh1b = (const float*)d_in[16];
  const float* w1    = (const float*)d_in[17];
  const float* w2    = (const float*)d_in[18];
  float* out = (float*)d_out;

  // workspace layout (floats): mean[8192] | y0[65536] | hlast[1024] | scale[128]
  float* ws    = (float*)d_ws;
  float* mean  = ws;
  float* y0    = ws + 8192;
  float* hlast = y0 + 65536;
  float* scale = hlast + 1024;

  mean_kernel<<<NB * TT, 256, 0, stream>>>(x, mean);
  lstm_l0_kernel<<<16, 32, 0, stream>>>(mean, wih0f, whh0f, bih0f, bhh0f,
                                        wih0b, whh0b, bih0b, bhh0b, y0);
  lstm_l1_kernel<<<16, 32, 0, stream>>>(y0, wih1f, whh1f, bih1f, bhh1f,
                                        wih1b, whh1b, bih1b, bhh1b, hlast);
  se_kernel<<<1, 32, 0, stream>>>(hlast, w1, w2, scale);
  scale_kernel<<<8192, 256, 0, stream>>>(x, scale, out);
}
